// DenseQAPNet_40724879900677
// MI455X (gfx1250) — matrix operand order repaired and checked by reading, not compiled, not run on hardware
//
#include <hip/hip_runtime.h>
#include <hip/hip_bf16.h>
#include <stdint.h>

// ---------------------------------------------------------------------------
// DenseQAPNet on MI455X (gfx1250): all dense work through v_wmma_f32_16x16x32_bf16
// (f32 accumulate). Edge MLPs fused with the neighbor-sum. Weights pre-swizzled
// into WMMA B-fragment layout; the edge kernel stages both weight matrices into
// (dynamic) LDS once per block -- CDNA5's 320KB WGP LDS -- and feeds WMMA with
// ds_load_b128. Per-wave LDS staging for the D->A fragment transpose between
// MLP layers, guarded by s_wait_dscnt. global_prefetch for next-iteration rows.
// Param indexing assumes jax tree_flatten order (dict keys sorted).
// ---------------------------------------------------------------------------

typedef __attribute__((ext_vector_type(16))) __bf16          v16bf;
typedef __attribute__((ext_vector_type(16))) unsigned short  v16u;
typedef __attribute__((ext_vector_type(8)))  unsigned short  v8u;
typedef __attribute__((ext_vector_type(8)))  float           v8f;

__device__ __forceinline__ unsigned short f2bf(float f) {
    unsigned u = __builtin_bit_cast(unsigned, f);
    unsigned r = 0x7FFFu + ((u >> 16) & 1u);          // round-to-nearest-even
    return (unsigned short)((u + r) >> 16);
}
__device__ __forceinline__ float bf2f(unsigned short h) {
    unsigned u = ((unsigned)h) << 16;
    return __builtin_bit_cast(float, u);
}
__device__ __forceinline__ float leakyf(float x) { return x >= 0.f ? x : 0.01f * x; }

__device__ __forceinline__ v8f wmma_bf16(v16u a, v16u b, v8f c) {
    return __builtin_amdgcn_wmma_f32_16x16x32_bf16(
        false, __builtin_bit_cast(v16bf, a),
        false, __builtin_bit_cast(v16bf, b),
        (short)0, c, false, false);
}

#define WAIT_DS() asm volatile("s_wait_dscnt 0" ::: "memory")

// A-fragment (16x32 bf16) from an LDS row-major [16][ldK] bf16 tile.
// Lane layout per ISA: lanes 0-15: K = base+{0..7,16..23}; lanes 16-31: +8.
__device__ __forceinline__ v16u lds_afrag(const unsigned short* row, int kk, int lane) {
    int base = kk * 32 + ((lane & 16) ? 8 : 0);
    v8u lo = *(const v8u*)(row + base);        // ds_load_b128
    v8u hi = *(const v8u*)(row + base + 16);   // ds_load_b128
    v16u a;
#pragma unroll
    for (int e = 0; e < 8; ++e) { a[e] = lo[e]; a[e + 8] = hi[e]; }
    return a;
}

// B-fragment (32x16 bf16) from a pre-swizzled fragment buffer (global or LDS):
// fragIdx -> 32 lanes x 16 bf16 contiguous (32B per lane -> b128 x2 loads).
__device__ __forceinline__ v16u frag16(const unsigned short* wf, int fragIdx, int lane) {
    return *(const v16u*)(wf + (((size_t)fragIdx * 32 + lane) << 4));
}

// ---------------------------------------------------------------------------
// Weight [K,N] f32 row-major -> bf16 B-fragment layout.
// (K_,N_): nt=N_/16, kk=K_/32, lane=(N_%16)+16*((K_%16)/8), e=(K_%8)+8*((K_%32)/16).
// ---------------------------------------------------------------------------
__global__ void prep_frag_kernel(const float* __restrict__ w, unsigned short* __restrict__ out,
                                 int K, int N) {
    int idx = blockIdx.x * 256 + threadIdx.x;
    if (idx >= K * N) return;
    int k = idx / N, n = idx - k * N;
    int nt = n >> 4;
    int kk = k >> 5;
    int lane = (n & 15) + (((k >> 3) & 1) << 4);
    int e = (k & 7) + (((k >> 4) & 1) << 3);
    int ksteps = K >> 5;
    out[(((size_t)(nt * ksteps + kk) * 32 + lane) << 4) + e] = f2bf(w[idx]);
}

// ---------------------------------------------------------------------------
// Generic node-level GEMM: Y = act( ((X1-mean)*rstd + X2 + X3) @ W + b ),
// M=512, K=N=128. One wave per 16 rows, 4 waves per block, 8 blocks.
// ---------------------------------------------------------------------------
__global__ __launch_bounds__(128) void node_gemm_kernel(
    const float* __restrict__ X1, const float* __restrict__ X2,
    const float* __restrict__ X3, const float* __restrict__ stats,
    const unsigned short* __restrict__ Wf, const float* __restrict__ bias,
    float* __restrict__ Y, int act)
{
    __shared__ unsigned short xs[4][16][128];
    const int lane = threadIdx.x & 31;
    const int wave = threadIdx.x >> 5;
    const int r0 = blockIdx.x * 64 + wave * 16;
    float mean = 0.f, rstd = 1.f;
    if (stats) { mean = stats[0]; rstd = stats[1]; }
    unsigned short (*tile)[128] = xs[wave];
    for (int idx = lane; idx < 16 * 128; idx += 32) {
        int r = idx >> 7, k = idx & 127;
        int g = (r0 + r) * 128 + k;
        float v = (X1[g] - mean) * rstd;
        if (X2) v += X2[g];
        if (X3) v += X3[g];
        tile[r][k] = f2bf(v);
    }
    WAIT_DS();
    const int m = lane & 15;
    const int roff = (lane & 16) ? 8 : 0;
    v16u a[4];
#pragma unroll
    for (int kk = 0; kk < 4; ++kk) a[kk] = lds_afrag(&tile[m][0], kk, lane);
#pragma unroll
    for (int nt = 0; nt < 8; ++nt) {
        v16u bf[4];
#pragma unroll
        for (int kk = 0; kk < 4; ++kk) bf[kk] = frag16(Wf, nt * 4 + kk, lane);
        float bv = bias[nt * 16 + m];
        v8f c;
#pragma unroll
        for (int r = 0; r < 8; ++r) c[r] = bv;
#pragma unroll
        for (int kk = 0; kk < 4; ++kk) c = wmma_bf16(a[kk], bf[kk], c);
#pragma unroll
        for (int r = 0; r < 8; ++r) {
            float v = c[r];
            if (act) v = leakyf(v);
            Y[(r0 + roff + r) * 128 + nt * 16 + m] = v;
        }
    }
}

// ---------------------------------------------------------------------------
// Fused edge conv: for a tile of 16 rows i, loop all j:
//   h1 = leaky(E[i,j]*w0row0 + t[j])   (t = nb@W0[1:]+b0, or b0 if no neighbors)
//   h2 = leaky(h1@W1+b1); h3 = leaky(h2@W2+b2); acc[i] += h3
// 4 waves split j; cross-wave reduce via LDS. W1/W2 fragment buffers staged in
// dynamic LDS once per block (113KB/block -- CDNA5 WGP has 320KB LDS).
// ---------------------------------------------------------------------------
#define EC_ACCS   0         // 4*16*128 f32   = 32768 B
#define EC_W1S    32768     // 128*128 bf16   = 32768 B
#define EC_W2S    65536     // 128*128 bf16   = 32768 B
#define EC_H2S    98304     // 4*16*128 bf16  = 16384 B
#define EC_W0S    114688    // 128 f32        =   512 B
#define EC_B0S    115200    // 128 f32        =   512 B
#define EC_SMEM   115712

__global__ __launch_bounds__(128) void edge_conv_kernel(
    const float* __restrict__ E, int transE,
    const float* __restrict__ t, const float* __restrict__ b0,
    const float* __restrict__ w0row0,
    const unsigned short* __restrict__ W1f, const float* __restrict__ b1,
    const unsigned short* __restrict__ W2f, const float* __restrict__ b2,
    float* __restrict__ xsum)
{
    extern __shared__ unsigned char smem[];
    float*          accs = (float*)(smem + EC_ACCS);
    unsigned short* w1s  = (unsigned short*)(smem + EC_W1S);
    unsigned short* w2s  = (unsigned short*)(smem + EC_W2S);
    unsigned short* h2s  = (unsigned short*)(smem + EC_H2S);
    float*          w0s  = (float*)(smem + EC_W0S);
    float*          b0s  = (float*)(smem + EC_B0S);

    const int lane = threadIdx.x & 31;
    const int wave = threadIdx.x >> 5;
    const int i0 = blockIdx.x * 16;

    {   // stage weight fragments into LDS (reused for all 512 j iterations)
        const uint4* s1 = (const uint4*)W1f;
        const uint4* s2 = (const uint4*)W2f;
        uint4* d1 = (uint4*)w1s;
        uint4* d2 = (uint4*)w2s;
        for (int idx = threadIdx.x; idx < 2048; idx += 128) { d1[idx] = s1[idx]; d2[idx] = s2[idx]; }
        for (int k = threadIdx.x; k < 128; k += 128) { w0s[k] = w0row0[k]; b0s[k] = b0[k]; }
    }
    __syncthreads();

    const int m = lane & 15;
    const int roff = (lane & 16) ? 8 : 0;
    v8f acc[8];
#pragma unroll
    for (int nt = 0; nt < 8; ++nt)
#pragma unroll
        for (int r = 0; r < 8; ++r) acc[nt][r] = 0.f;
    unsigned short* tile = h2s + wave * 16 * 128;

    for (int j = wave; j < 512; j += 4) {
        // prefetch next iteration's rows (global_prefetch_b8)
        int jn = j + 4;
        if (jn < 512) {
            if (t) __builtin_prefetch(t + jn * 128, 0, 0);
            __builtin_prefetch(transE ? (E + jn * 512 + i0) : (E + (i0 + m) * 512 + jn), 0, 0);
        }
        float e = transE ? E[j * 512 + (i0 + m)] : E[(i0 + m) * 512 + j];
        const float* tj = t ? (t + j * 128) : nullptr;
        v16u a1[4];
#pragma unroll
        for (int kk = 0; kk < 4; ++kk) {
#pragma unroll
            for (int q = 0; q < 16; ++q) {
                int k = kk * 32 + roff + ((q < 8) ? q : (q + 8));
                float tv = tj ? tj[k] : b0s[k];
                a1[kk][q] = f2bf(leakyf(e * w0s[k] + tv));
            }
        }
#pragma unroll
        for (int nt = 0; nt < 8; ++nt) {
            v16u bf[4];
#pragma unroll
            for (int kk = 0; kk < 4; ++kk) bf[kk] = frag16(w1s, nt * 4 + kk, lane);
            float bv = b1[nt * 16 + m];
            v8f c;
#pragma unroll
            for (int r = 0; r < 8; ++r) c[r] = bv;
#pragma unroll
            for (int kk = 0; kk < 4; ++kk) c = wmma_bf16(a1[kk], bf[kk], c);
#pragma unroll
            for (int r = 0; r < 8; ++r) tile[(roff + r) * 128 + nt * 16 + m] = f2bf(leakyf(c[r]));
        }
        WAIT_DS();  // per-wave D->A transpose handoff through LDS
        v16u a2[4];
#pragma unroll
        for (int kk = 0; kk < 4; ++kk) a2[kk] = lds_afrag(tile + m * 128, kk, lane);
#pragma unroll
        for (int nt = 0; nt < 8; ++nt) {
            v16u bf[4];
#pragma unroll
            for (int kk = 0; kk < 4; ++kk) bf[kk] = frag16(w2s, nt * 4 + kk, lane);
            float bv = b2[nt * 16 + m];
            v8f c;
#pragma unroll
            for (int r = 0; r < 8; ++r) c[r] = bv;
#pragma unroll
            for (int kk = 0; kk < 4; ++kk) c = wmma_bf16(a2[kk], bf[kk], c);
#pragma unroll
            for (int r = 0; r < 8; ++r) acc[nt][r] += leakyf(c[r]);
        }
    }
#pragma unroll
    for (int nt = 0; nt < 8; ++nt)
#pragma unroll
        for (int r = 0; r < 8; ++r)
            accs[(wave * 16 + roff + r) * 128 + nt * 16 + m] = acc[nt][r];
    __syncthreads();
    for (int idx = threadIdx.x; idx < 16 * 128; idx += 128) {
        int r = idx >> 7, k = idx & 127;
        xsum[(i0 + r) * 128 + k] =
            accs[(0 * 16 + r) * 128 + k] + accs[(1 * 16 + r) * 128 + k] +
            accs[(2 * 16 + r) * 128 + k] + accs[(3 * 16 + r) * 128 + k];
    }
}

// Global layer-norm stats over the whole [512,128] tensor -> {mean, rsqrt(var+eps)}
__global__ void reduce_stats_kernel(const float* __restrict__ X, int n, float* __restrict__ out) {
    __shared__ float ss[256];
    __shared__ float sq[256];
    float s = 0.f, q = 0.f;
    for (int idx = threadIdx.x; idx < n; idx += 256) { float v = X[idx]; s += v; q += v * v; }
    ss[threadIdx.x] = s; sq[threadIdx.x] = q;
    __syncthreads();
    for (int o = 128; o > 0; o >>= 1) {
        if ((int)threadIdx.x < o) {
            ss[threadIdx.x] += ss[threadIdx.x + o];
            sq[threadIdx.x] += sq[threadIdx.x + o];
        }
        __syncthreads();
    }
    if (threadIdx.x == 0) {
        float inv = 1.f / (float)n;
        float mean = ss[0] * inv;
        float var = sq[0] * inv - mean * mean;
        out[0] = mean;
        out[1] = rsqrtf(var + 1e-5f);
    }
}

// Per-node sum / sumsq for the pair LayerNorm
__global__ void node_stats_kernel(const float* __restrict__ a, const float* __restrict__ b,
                                  float* __restrict__ suma, float* __restrict__ sqa,
                                  float* __restrict__ sumb, float* __restrict__ sqb) {
    int i = blockIdx.x * 128 + threadIdx.x;
    if (i >= 512) return;
    float s = 0.f, q = 0.f;
    for (int k = 0; k < 128; ++k) { float v = a[i * 128 + k]; s += v; q += v * v; }
    suma[i] = s; sqa[i] = q;
    s = 0.f; q = 0.f;
    for (int k = 0; k < 128; ++k) { float v = b[i * 128 + k]; s += v; q += v * v; }
    sumb[i] = s; sqb[i] = q;
}

// ---------------------------------------------------------------------------
// Fused link head: per 16-pair tile (one i, 16 consecutive j):
// per-pair LN(256) w/ affine -> 3x leaky linears (WMMA) -> final dot(128)+bias.
// ---------------------------------------------------------------------------
__global__ __launch_bounds__(128) void link_kernel(
    const float* __restrict__ a, const float* __restrict__ b,
    const float* __restrict__ suma, const float* __restrict__ sqa,
    const float* __restrict__ sumb, const float* __restrict__ sqb,
    const float* __restrict__ gamma, const float* __restrict__ beta,
    const unsigned short* __restrict__ W0f, const float* __restrict__ bias0,
    const unsigned short* __restrict__ W1f, const float* __restrict__ bias1,
    const unsigned short* __restrict__ W2f, const float* __restrict__ bias2,
    const float* __restrict__ w3, const float* __restrict__ b3,
    float* __restrict__ out)
{
    __shared__ unsigned short x0s[4][16][256];
    __shared__ unsigned short hs[4][16][128];
    __shared__ float w3s[128];
    __shared__ float pm[4][16];
    __shared__ float pr[4][16];
    const int lane = threadIdx.x & 31;
    const int wave = threadIdx.x >> 5;
    for (int k = threadIdx.x; k < 128; k += 128) w3s[k] = w3[k];
    const int p0 = (blockIdx.x * 4 + wave) * 16;
    const int i = p0 >> 9;
    const int j0 = p0 & 511;
    if (lane < 16) {
        int j = j0 + lane;
        float mean = (suma[i] + sumb[j]) * (1.f / 256.f);
        float var = (sqa[i] + sqb[j]) * (1.f / 256.f) - mean * mean;
        pm[wave][lane] = mean;
        pr[wave][lane] = rsqrtf(var + 1e-5f);
    }
    __syncthreads();
    unsigned short (*t0)[256] = x0s[wave];
    for (int idx = lane; idx < 16 * 256; idx += 32) {
        int mm = idx >> 8, k = idx & 255;
        int j = j0 + mm;
        float f = (k < 128) ? a[i * 128 + k] : b[j * 128 + (k - 128)];
        float v = (f - pm[wave][mm]) * pr[wave][mm] * gamma[k] + beta[k];
        t0[mm][k] = f2bf(v);
    }
    WAIT_DS();
    const int m = lane & 15;
    const int roff = (lane & 16) ? 8 : 0;
    unsigned short (*th)[128] = hs[wave];
    {   // GEMM1: K=256
        v16u a0[8];
#pragma unroll
        for (int kk = 0; kk < 8; ++kk) a0[kk] = lds_afrag(&t0[m][0], kk, lane);
#pragma unroll
        for (int nt = 0; nt < 8; ++nt) {
            v16u bf[8];
#pragma unroll
            for (int kk = 0; kk < 8; ++kk) bf[kk] = frag16(W0f, nt * 8 + kk, lane);
            float bv = bias0[nt * 16 + m];
            v8f c;
#pragma unroll
            for (int r = 0; r < 8; ++r) c[r] = bv;
#pragma unroll
            for (int kk = 0; kk < 8; ++kk) c = wmma_bf16(a0[kk], bf[kk], c);
#pragma unroll
            for (int r = 0; r < 8; ++r) th[roff + r][nt * 16 + m] = f2bf(leakyf(c[r]));
        }
    }
    WAIT_DS();
    {   // GEMM2
        v16u aa[4];
#pragma unroll
        for (int kk = 0; kk < 4; ++kk) aa[kk] = lds_afrag(&th[m][0], kk, lane);
#pragma unroll
        for (int nt = 0; nt < 8; ++nt) {
            v16u bf[4];
#pragma unroll
            for (int kk = 0; kk < 4; ++kk) bf[kk] = frag16(W1f, nt * 4 + kk, lane);
            float bv = bias1[nt * 16 + m];
            v8f c;
#pragma unroll
            for (int r = 0; r < 8; ++r) c[r] = bv;
#pragma unroll
            for (int kk = 0; kk < 4; ++kk) c = wmma_bf16(aa[kk], bf[kk], c);
#pragma unroll
            for (int r = 0; r < 8; ++r) th[roff + r][nt * 16 + m] = f2bf(leakyf(c[r]));
        }
    }
    WAIT_DS();
    {   // GEMM3
        v16u aa[4];
#pragma unroll
        for (int kk = 0; kk < 4; ++kk) aa[kk] = lds_afrag(&th[m][0], kk, lane);
#pragma unroll
        for (int nt = 0; nt < 8; ++nt) {
            v16u bf[4];
#pragma unroll
            for (int kk = 0; kk < 4; ++kk) bf[kk] = frag16(W2f, nt * 4 + kk, lane);
            float bv = bias2[nt * 16 + m];
            v8f c;
#pragma unroll
            for (int r = 0; r < 8; ++r) c[r] = bv;
#pragma unroll
            for (int kk = 0; kk < 4; ++kk) c = wmma_bf16(aa[kk], bf[kk], c);
#pragma unroll
            for (int r = 0; r < 8; ++r) th[roff + r][nt * 16 + m] = f2bf(leakyf(c[r]));
        }
    }
    WAIT_DS();
    if (lane < 16) {
        float dot = b3[0];
        for (int k = 0; k < 128; ++k) dot += bf2f(th[lane][k]) * w3s[k];
        out[i * 512 + j0 + lane] = dot;
    }
}

// ---------------------------------------------------------------------------
// Host orchestration.
// Param leaf order (jax tree_flatten: dict keys sorted): A,B,L, then for each of
// a0,a1,a2,b0,b1,b2 the keys comb,l_enc,l_tr,q_enc,q_tr (each mlp as w,b pairs),
// then link (4 linears), then pair_norm (gamma,beta).
// ---------------------------------------------------------------------------
extern "C" void kernel_launch(void* const* d_in, const int* in_sizes, int n_in,
                              void* d_out, int out_size, void* d_ws, size_t ws_size,
                              hipStream_t stream)
{
    (void)in_sizes; (void)n_in; (void)out_size; (void)ws_size;
    const float* A = (const float*)d_in[0];
    const float* B = (const float*)d_in[1];
    const float* L = (const float*)d_in[2];
    auto P = [&](int i) { return (const float*)d_in[i]; };

    size_t off = 0;
    auto alloc = [&](size_t bytes) {
        void* p = (char*)d_ws + off;
        off = (off + bytes + 255) & ~(size_t)255;
        return p;
    };
    auto allocF = [&](size_t n) { return (float*)alloc(n * 4); };
    auto allocU = [&](size_t n) { return (unsigned short*)alloc(n * 2); };
    auto prep = [&](const float* w, int K, int Nn) {
        unsigned short* f = allocU((size_t)K * Nn);
        int total = K * Nn;
        prep_frag_kernel<<<dim3((total + 255) / 256), dim3(256), 0, stream>>>(w, f, K, Nn);
        return f;
    };

    float* aBuf[2] = { allocF(512 * 128), allocF(512 * 128) };
    float* bBuf[2] = { allocF(512 * 128), allocF(512 * 128) };
    float* xsq   = allocF(512 * 128);
    float* xsl   = allocF(512 * 128);
    float* tbuf  = allocF(512 * 128);
    float* tmp1  = allocF(512 * 128);
    float* tmp2  = allocF(512 * 128);
    float* tmp3  = allocF(512 * 128);
    float* statsQ = allocF(8);
    float* statsL = allocF(8);
    float* suma = allocF(512); float* sqa = allocF(512);
    float* sumb = allocF(512); float* sqb = allocF(512);

    const float* aOld = nullptr;
    const float* bOld = nullptr;

    for (int l = 0; l < 3; ++l) {
        const float* newDst[2] = { nullptr, nullptr };
        for (int br = 0; br < 2; ++br) {
            int base = 3 + (br * 3 + l) * 22;
            const float* cw[3]  = { P(base + 0),  P(base + 2),  P(base + 4) };
            const float* cb[3]  = { P(base + 1),  P(base + 3),  P(base + 5) };
            const float* le_w0  = P(base + 6);   const float* le_b0 = P(base + 7);
            const float* le_w1  = P(base + 8);   const float* le_b1 = P(base + 9);
            const float* le_w2  = P(base + 10);  const float* le_b2 = P(base + 11);
            const float* ltr_w  = P(base + 12);  const float* ltr_b = P(base + 13);
            const float* qe_w0  = P(base + 14);  const float* qe_b0 = P(base + 15);
            const float* qe_w1  = P(base + 16);  const float* qe_b1 = P(base + 17);
            const float* qe_w2  = P(base + 18);  const float* qe_b2 = P(base + 19);
            const float* qtr_w  = P(base + 20);  const float* qtr_b = P(base + 21);

            unsigned short* qe_w1f = prep(qe_w1, 128, 128);
            unsigned short* qe_w2f = prep(qe_w2, 128, 128);
            unsigned short* le_w1f = prep(le_w1, 128, 128);
            unsigned short* le_w2f = prep(le_w2, 128, 128);
            unsigned short* qtrf   = prep(qtr_w, 128, 128);
            unsigned short* ltrf   = prep(ltr_w, 128, 128);
            unsigned short* cwf[3] = { prep(cw[0], 128, 128), prep(cw[1], 128, 128),
                                       prep(cw[2], 128, 128) };
            unsigned short* qe_w0sf = nullptr;
            unsigned short* le_w0sf = nullptr;
            if (l > 0) {
                qe_w0sf = prep(qe_w0 + 128, 128, 128);  // rows 1..128 of [129,128]
                le_w0sf = prep(le_w0 + 128, 128, 128);
            }

            const float* qE  = (br == 0) ? A : B;
            const float* qNb = (br == 0) ? aOld : bOld;
            const float* lNb = (br == 0) ? bOld : aOld;
            int lTrans = (br == 0) ? 0 : 1;

            // q-conv
            if (qNb) {
                node_gemm_kernel<<<dim3(8), dim3(128), 0, stream>>>(
                    qNb, nullptr, nullptr, nullptr, qe_w0sf, qe_b0, tbuf, 0);
                edge_conv_kernel<<<dim3(32), dim3(128), EC_SMEM, stream>>>(
                    qE, 0, tbuf, qe_b0, qe_w0, qe_w1f, qe_b1, qe_w2f, qe_b2, xsq);
            } else {
                edge_conv_kernel<<<dim3(32), dim3(128), EC_SMEM, stream>>>(
                    qE, 0, nullptr, qe_b0, qe_w0, qe_w1f, qe_b1, qe_w2f, qe_b2, xsq);
            }
            reduce_stats_kernel<<<dim3(1), dim3(256), 0, stream>>>(xsq, 512 * 128, statsQ);

            // l-conv
            if (lNb) {
                node_gemm_kernel<<<dim3(8), dim3(128), 0, stream>>>(
                    lNb, nullptr, nullptr, nullptr, le_w0sf, le_b0, tbuf, 0);
                edge_conv_kernel<<<dim3(32), dim3(128), EC_SMEM, stream>>>(
                    L, lTrans, tbuf, le_b0, le_w0, le_w1f, le_b1, le_w2f, le_b2, xsl);
            } else {
                edge_conv_kernel<<<dim3(32), dim3(128), EC_SMEM, stream>>>(
                    L, lTrans, nullptr, le_b0, le_w0, le_w1f, le_b1, le_w2f, le_b2, xsl);
            }
            reduce_stats_kernel<<<dim3(1), dim3(256), 0, stream>>>(xsl, 512 * 128, statsL);

            // transforms (global-LN applied via stats), then comb MLP
            node_gemm_kernel<<<dim3(8), dim3(128), 0, stream>>>(
                xsq, nullptr, nullptr, statsQ, qtrf, qtr_b, tmp1, 1);
            node_gemm_kernel<<<dim3(8), dim3(128), 0, stream>>>(
                xsl, nullptr, nullptr, statsL, ltrf, ltr_b, tmp2, 1);
            const float* self = (br == 0) ? aOld : bOld;
            node_gemm_kernel<<<dim3(8), dim3(128), 0, stream>>>(
                tmp1, tmp2, self, nullptr, cwf[0], cb[0], tmp3, 1);
            node_gemm_kernel<<<dim3(8), dim3(128), 0, stream>>>(
                tmp3, nullptr, nullptr, nullptr, cwf[1], cb[1], tmp1, 1);
            float* dst = (br == 0) ? aBuf[l & 1] : bBuf[l & 1];
            node_gemm_kernel<<<dim3(8), dim3(128), 0, stream>>>(
                tmp1, nullptr, nullptr, nullptr, cwf[2], cb[2], dst, 1);
            newDst[br] = dst;
        }
        aOld = newDst[0];
        bOld = newDst[1];
    }

    // link head
    unsigned short* W0f = prep(P(135), 256, 128);
    const float* lb0 = P(136);
    unsigned short* W1f = prep(P(137), 128, 128);
    const float* lb1 = P(138);
    unsigned short* W2f = prep(P(139), 128, 128);
    const float* lb2 = P(140);
    const float* w3 = P(141);
    const float* b3 = P(142);
    const float* gamma = P(143);
    const float* beta  = P(144);

    node_stats_kernel<<<dim3(4), dim3(128), 0, stream>>>(aOld, bOld, suma, sqa, sumb, sqb);
    link_kernel<<<dim3(4096), dim3(128), 0, stream>>>(
        aOld, bOld, suma, sqa, sumb, sqb, gamma, beta,
        W0f, lb0, W1f, lb1, W2f, lb2, w3, b3, (float*)d_out);
}
